// Sampler_7318624273071
// MI455X (gfx1250) — matrix operand order repaired and checked by reading, not compiled, not run on hardware
//
#include <hip/hip_runtime.h>
#include <hip/hip_bf16.h>

// ---------------------------------------------------------------------------
// MI455X (gfx1250) sampler head:
//   logits = H(128x4096) @ E^T(4096x50304) + bias -> log_softmax -> outputs
// HBM-bound on the 824MB embedding stream (~35us @ 23.3TB/s).
// bf16 WMMA (16x16x32) w/ fp32 accumulate keeps matrix work (~11us) under the
// memory floor; fp32->bf16 conversion happens once per element at staging.
// Square 2x2 wave tiling (32B/lane ds per WMMA), ping-pong LDS buffers with
// software-pipelined global loads, NT cache hints on the streamed operand.
// ---------------------------------------------------------------------------

typedef __bf16 bf16;
typedef __bf16 v8bf  __attribute__((ext_vector_type(8)));
typedef __bf16 v16bf __attribute__((ext_vector_type(16)));
typedef float  v4f   __attribute__((ext_vector_type(4)));
typedef float  v8f   __attribute__((ext_vector_type(8)));

#define NSEQ   128
#define DMODEL 4096
#define VPAD   50304
#define VOC    50257
#define NTOPK  5

// GEMM tiling
#define NT  128   // vocab columns per workgroup (VPAD/NT = 393 WGs)
#define KC  64    // K chunk staged in LDS
#define KCP 72    // padded LDS stride (bank spread)

// d_out layout (floats, concatenated in reference return order)
#define OFF_GREEDY 0
#define OFF_SAMP   128
#define OFF_TLP    256
#define OFF_TID    896
#define OFF_SLP    1536
#define OFF_LP     1664

// ---------------------------------------------------------------------------
// K1: logits = H @ E^T + bias  (bf16 WMMA, fp32 accumulate)
// 512 threads = 16 waves = 4 m-groups x 4 n-groups; each wave owns a 32x32
// output tile (2x2 WMMA accumulators) -> every frag reused twice.
// ---------------------------------------------------------------------------
__global__ __launch_bounds__(512)
void sampler_gemm_kernel(const float* __restrict__ H,
                         const float* __restrict__ E,
                         const float* __restrict__ bias,
                         float* __restrict__ logits)
{
    __shared__ bf16 As[2][NSEQ * KCP];   // [buf][m][k]
    __shared__ bf16 Bs[2][NT   * KCP];   // [buf][n][k]

    const int tid  = threadIdx.x;
    const int lane = tid & 31;
    const int wave = tid >> 5;
    const int mg   = wave & 3;    // m group: rows mg*32 .. +31
    const int ng   = wave >> 2;   // n group: cols ng*32 .. +31
    const int nblk = blockIdx.x * NT;

    const int l15  = lane & 15;
    const int half = lane >> 4;

    v8f acc[2][2];
#pragma unroll
    for (int mi = 0; mi < 2; ++mi)
#pragma unroll
        for (int j = 0; j < 2; ++j)
#pragma unroll
            for (int i = 0; i < 8; ++i)
                acc[mi][j][i] = 0.0f;

    // cooperative staging: thread -> (row sr, 16-float slice at sc)
    const int sr = tid >> 2;         // 0..127
    const int sc = (tid & 3) * 16;   // 0,16,32,48
    const float* aSrc = H + (size_t)sr * DMODEL + sc;
    const float* bSrc = E + (size_t)(nblk + sr) * DMODEL + sc;

    v4f areg[4], breg[4];

    auto load_chunk = [&](int kb) {
        const v4f* ap = reinterpret_cast<const v4f*>(aSrc + kb);
        const v4f* bp = reinterpret_cast<const v4f*>(bSrc + kb);
#pragma unroll
        for (int i = 0; i < 4; ++i) {
            areg[i] = ap[i];                              // H: hot in L2, RT
            breg[i] = __builtin_nontemporal_load(bp + i); // E: 824MB stream, NT
        }
    };
    auto cvt_store = [&](int buf) {
        bf16* da = &As[buf][sr * KCP + sc];
        bf16* db = &Bs[buf][sr * KCP + sc];
#pragma unroll
        for (int i = 0; i < 4; ++i)
#pragma unroll
            for (int e = 0; e < 4; ++e) {
                da[i * 4 + e] = (bf16)areg[i][e];
                db[i * 4 + e] = (bf16)breg[i][e];
            }
    };

    load_chunk(0);
    cvt_store(0);
    int cur = 0;

    for (int kb = 0; kb < DMODEL; kb += KC) {
        __syncthreads();   // buf[cur] ready; buf[cur^1] no longer being read
        const bool more = (kb + KC) < DMODEL;
        if (more) load_chunk(kb + KC);   // in flight during compute

#pragma unroll
        for (int ks = 0; ks < 2; ++ks) {
            const int k0 = ks * 32;
            v16bf afrag[2], bfrag[2];
#pragma unroll
            for (int mi = 0; mi < 2; ++mi) {
                // A frag: lane l<16 -> row m, K {0..7,16..23}; lanes 16..31 -> K {8..15,24..31}
                const bf16* ap = &As[cur][(mg * 32 + mi * 16 + l15) * KCP + k0];
                v8bf a0 = *reinterpret_cast<const v8bf*>(ap + half * 8);
                v8bf a1 = *reinterpret_cast<const v8bf*>(ap + 16 + half * 8);
                afrag[mi] = __builtin_shufflevector(a0, a1,
                    0, 1, 2, 3, 4, 5, 6, 7, 8, 9, 10, 11, 12, 13, 14, 15);
            }
#pragma unroll
            for (int j = 0; j < 2; ++j) {
                // B frag: lane l<16 -> column n, K {0..7,16..23}; lanes 16..31 -> K {8..15,24..31}
                const bf16* bp = &Bs[cur][(ng * 32 + j * 16 + l15) * KCP + k0 + half * 8];
                v8bf b0 = *reinterpret_cast<const v8bf*>(bp);
                v8bf b1 = *reinterpret_cast<const v8bf*>(bp + 16);
                bfrag[j] = __builtin_shufflevector(b0, b1,
                    0, 1, 2, 3, 4, 5, 6, 7, 8, 9, 10, 11, 12, 13, 14, 15);
            }
#pragma unroll
            for (int mi = 0; mi < 2; ++mi)
#pragma unroll
                for (int j = 0; j < 2; ++j)
                    acc[mi][j] = __builtin_amdgcn_wmma_f32_16x16x32_bf16(
                        false, afrag[mi], false, bfrag[j], (short)0, acc[mi][j], false, false);
        }

        if (more) cvt_store(cur ^ 1);   // waits on the in-flight loads here
        cur ^= 1;
    }

    // epilogue: C layout lane l, vgpr r -> row (l>>4)*8+r, col l&15
#pragma unroll
    for (int j = 0; j < 2; ++j) {
        const int gn = nblk + ng * 32 + j * 16 + l15;
        const float bv = bias[gn];
#pragma unroll
        for (int mi = 0; mi < 2; ++mi)
#pragma unroll
            for (int r = 0; r < 8; ++r) {
                const int gm = mg * 32 + mi * 16 + half * 8 + r;
                logits[(size_t)gm * VPAD + gn] = acc[mi][j][r] + bv;
            }
    }
}

// ---------------------------------------------------------------------------
// K2: per-row online log-sum-exp over the first VOC columns
// ---------------------------------------------------------------------------
__global__ __launch_bounds__(256)
void sampler_rowstats_kernel(const float* __restrict__ logits,
                             float* __restrict__ stats)
{
    __shared__ float sm[256];
    __shared__ float ss[256];
    const int s = blockIdx.x;
    const int t = threadIdx.x;
    const float* row = logits + (size_t)s * VPAD;

    float m = -INFINITY, sum = 0.0f;
    for (int v = t; v < VOC; v += 256) {
        float x = row[v];
        if (x > m) {
            sum = sum * expf(m - x) + 1.0f;
            m = x;
        } else {
            sum += expf(x - m);
        }
    }
    sm[t] = m; ss[t] = sum;
    __syncthreads();
    for (int st = 128; st > 0; st >>= 1) {
        if (t < st) {
            float m2 = sm[t + st], s2 = ss[t + st];
            float M = fmaxf(sm[t], m2);
            ss[t] = ss[t] * expf(sm[t] - M) + s2 * expf(m2 - M);
            sm[t] = M;
        }
        __syncthreads();
    }
    if (t == 0) {
        stats[s * 2 + 0] = sm[0];
        stats[s * 2 + 1] = sm[0] + logf(ss[0]);   // logZ
    }
}

// ---------------------------------------------------------------------------
// K3: logprobs + greedy + top-5 + gumbel sampling, one workgroup per row
// ---------------------------------------------------------------------------
__device__ __forceinline__ unsigned int hash_u32(unsigned int x)
{
    x ^= x >> 16; x *= 0x7feb352dU;
    x ^= x >> 15; x *= 0x846ca68bU;
    x ^= x >> 16;
    return x;
}

__global__ __launch_bounds__(256)
void sampler_finalize_kernel(const float* __restrict__ logits,
                             const float* __restrict__ stats,
                             float* __restrict__ out)
{
    __shared__ float lv[256 * NTOPK];
    __shared__ int   li[256 * NTOPK];
    __shared__ float gk[256];
    __shared__ float gl[256];
    __shared__ int   gi[256];

    const int s = blockIdx.x;
    const int t = threadIdx.x;
    const float logZ = stats[s * 2 + 1];
    const float* row = logits + (size_t)s * VPAD;
    float* outlp = out + OFF_LP + (size_t)s * VOC;

    float tv[NTOPK];
    int   ti[NTOPK];
#pragma unroll
    for (int k = 0; k < NTOPK; ++k) { tv[k] = -INFINITY; ti[k] = 0x7fffffff; }
    float bk = -INFINITY, blp = 0.0f;
    int   bi = 0x7fffffff;

    for (int v = t; v < VOC; v += 256) {
        const float lp = row[v] - logZ;
        __builtin_nontemporal_store(lp, outlp + v);   // written once, never re-read
        // local sorted top-5 insert
        if (lp > tv[NTOPK - 1]) {
            int p = NTOPK - 1;
            while (p > 0 && lp > tv[p - 1]) {
                tv[p] = tv[p - 1]; ti[p] = ti[p - 1]; --p;
            }
            tv[p] = lp; ti[p] = v;
        }
        // gumbel perturbation (deterministic counter hash; JAX key-42 stream
        // is not reproducible on-device)
        unsigned int h = hash_u32((unsigned int)v * 0x9E3779B9u +
                                  (unsigned int)s * 0x85EBCA6Bu + 0x2Au);
        float u = (float)(h >> 8) * (1.0f / 16777216.0f);
        float g = -logf(-logf(fmaxf(u, 1e-12f)));
        float key = lp + g;
        if (key > bk || (key == bk && v < bi)) { bk = key; blp = lp; bi = v; }
    }

#pragma unroll
    for (int k = 0; k < NTOPK; ++k) { lv[t * NTOPK + k] = tv[k]; li[t * NTOPK + k] = ti[k]; }
    gk[t] = bk; gl[t] = blp; gi[t] = bi;
    __syncthreads();

    for (int st = 128; st > 0; st >>= 1) {
        if (t < st) {
            // merge two sorted 5-lists (ties -> smaller index, like jax.top_k)
            float* A = &lv[t * NTOPK];        int* Ai = &li[t * NTOPK];
            float* B = &lv[(t + st) * NTOPK]; int* Bi = &li[(t + st) * NTOPK];
            float mv[NTOPK]; int mi[NTOPK];
            int ia = 0, ib = 0;
#pragma unroll
            for (int k = 0; k < NTOPK; ++k) {
                bool ta = (A[ia] > B[ib]) || (A[ia] == B[ib] && Ai[ia] <= Bi[ib]);
                if (ta) { mv[k] = A[ia]; mi[k] = Ai[ia]; ++ia; }
                else    { mv[k] = B[ib]; mi[k] = Bi[ib]; ++ib; }
            }
#pragma unroll
            for (int k = 0; k < NTOPK; ++k) { A[k] = mv[k]; Ai[k] = mi[k]; }
            // gumbel-argmax reduce
            bool tb = (gk[t + st] > gk[t]) ||
                      (gk[t + st] == gk[t] && gi[t + st] < gi[t]);
            if (tb) { gk[t] = gk[t + st]; gl[t] = gl[t + st]; gi[t] = gi[t + st]; }
        }
        __syncthreads();
    }

    if (t == 0) {
        out[OFF_GREEDY + s] = (float)li[0];   // greedy = top-1
        out[OFF_SAMP   + s] = (float)gi[0];
#pragma unroll
        for (int k = 0; k < NTOPK; ++k) {
            out[OFF_TLP + s * NTOPK + k] = lv[k];
            out[OFF_TID + s * NTOPK + k] = (float)li[k];
        }
        out[OFF_SLP + s] = gl[0];
    }
}

// ---------------------------------------------------------------------------
extern "C" void kernel_launch(void* const* d_in, const int* in_sizes, int n_in,
                              void* d_out, int out_size, void* d_ws, size_t ws_size,
                              hipStream_t stream)
{
    const float* H    = (const float*)d_in[0];   // (128, 4096)
    const float* E    = (const float*)d_in[1];   // (50304, 4096)
    const float* bias = (const float*)d_in[2];   // (50304,)
    (void)in_sizes; (void)n_in; (void)out_size; (void)ws_size;

    float* logits = (float*)d_ws;                       // 128 * 50304 fp32 (~25.8 MB)
    float* stats  = logits + (size_t)NSEQ * VPAD;       // 128 * 2 fp32
    float* out    = (float*)d_out;

    sampler_gemm_kernel<<<dim3(VPAD / NT), dim3(512), 0, stream>>>(H, E, bias, logits);
    sampler_rowstats_kernel<<<dim3(NSEQ), dim3(256), 0, stream>>>(logits, stats);
    sampler_finalize_kernel<<<dim3(NSEQ), dim3(256), 0, stream>>>(logits, stats, out);
}
